// ProductKeysMemory_60696477827387
// MI455X (gfx1250) — matrix operand order repaired and checked by reading, not compiled, not run on hardware
//
#include <hip/hip_runtime.h>
#include <hip/hip_bf16.h>
#include <math.h>

// Problem constants (fixed by the reference)
#define TDIM   2048      // B*S
#define DMODEL 512
#define HQDIM  2048      // H * QUERY_DIM
#define NHEAD  4
#define QDIM   512
#define QHALF  256
#define NSUB   512
#define KSEL   32
#define BNCHUNK 8        // row chunks for BN partial stats

typedef __attribute__((ext_vector_type(16))) __bf16 v16bf;
typedef __attribute__((ext_vector_type(8)))  float  v8f;

union Frag16 { v16bf v; unsigned short u[16]; };

__device__ __forceinline__ unsigned short f32_to_bf16_rne(float f) {
  unsigned int u = __float_as_uint(f);
  u += 0x7FFFu + ((u >> 16) & 1u);   // round to nearest even
  return (unsigned short)(u >> 16);
}

// A-matrix 16x32 bf16 fragment (ISA 7.12.2): lanes 0-15 -> K{k0..k0+7, k0+16..k0+23},
// lanes 16-31 -> K{k0+8..k0+15, k0+24..k0+31}; row = tileM + (lane&15).
__device__ __forceinline__ v16bf load_a_frag(const unsigned short* A, int lda,
                                             int row, int k0, int hi) {
  Frag16 f;
  const unsigned short* p = A + (size_t)row * lda + k0 + hi * 8;
#pragma unroll
  for (int i = 0; i < 8; i++) f.u[i] = p[i];
#pragma unroll
  for (int i = 0; i < 8; i++) f.u[8 + i] = p[16 + i];
  return f.v;
}

// B-matrix 32x16 bf16 fragment, B stored transposed (Bt[col][k], contiguous in k):
// lanes 0-15 hold K{k0..k0+7, k0+16..k0+23} for col=lane, lanes 16-31 hold K{k0+8..k0+15, k0+24..k0+31}.
__device__ __forceinline__ v16bf load_b_frag(const unsigned short* Bt, int ldb,
                                             int col, int k0, int hi) {
  Frag16 f;
  const unsigned short* p = Bt + (size_t)col * ldb + k0 + hi * 8;
#pragma unroll
  for (int i = 0; i < 8; i++) f.u[i] = p[i];
#pragma unroll
  for (int i = 0; i < 8; i++) f.u[8 + i] = p[16 + i];
  return f.v;
}

// ---------------- conversion kernels ----------------
__global__ void k_convert_bf16(const float* __restrict__ in,
                               unsigned short* __restrict__ out, int n) {
  int i = blockIdx.x * blockDim.x + threadIdx.x;
  if (i < n) out[i] = f32_to_bf16_rne(in[i]);
}

// Wq [DMODEL, HQDIM] -> wbT [HQDIM, DMODEL] (bf16), so GEMM B-frags are contiguous.
__global__ void k_transpose_bf16(const float* __restrict__ Wq,
                                 unsigned short* __restrict__ wbT) {
  int i = blockIdx.x * blockDim.x + threadIdx.x;
  if (i >= DMODEL * HQDIM) return;
  int k = i / HQDIM;
  int ncol = i % HQDIM;
  wbT[(size_t)ncol * DMODEL + k] = f32_to_bf16_rne(Wq[i]);
}

// ------- q projection GEMM: each wave computes a 16x64 strip (4 acc, A reused) -------
__global__ void k_gemm_q(const unsigned short* __restrict__ xb,
                         const unsigned short* __restrict__ wbT,
                         const float* __restrict__ bq,
                         float* __restrict__ qraw) {
  int wid = threadIdx.x >> 5;
  int lane = threadIdx.x & 31;
  int laneL = lane & 15, hi = lane >> 4;
  int tileN = (blockIdx.x * 4 + wid) * 64;     // 64-wide strip per wave
  int tileM = blockIdx.y * 16;
  v8f acc0 = {}, acc1 = {}, acc2 = {}, acc3 = {};
  for (int k0 = 0; k0 < DMODEL; k0 += 32) {
    v16bf a  = load_a_frag(xb, DMODEL, tileM + laneL, k0, hi);
    v16bf b0 = load_b_frag(wbT, DMODEL, tileN +  0 + laneL, k0, hi);
    v16bf b1 = load_b_frag(wbT, DMODEL, tileN + 16 + laneL, k0, hi);
    v16bf b2 = load_b_frag(wbT, DMODEL, tileN + 32 + laneL, k0, hi);
    v16bf b3 = load_b_frag(wbT, DMODEL, tileN + 48 + laneL, k0, hi);
    acc0 = __builtin_amdgcn_wmma_f32_16x16x32_bf16(false, a, false, b0, (short)0, acc0, false, false);
    acc1 = __builtin_amdgcn_wmma_f32_16x16x32_bf16(false, a, false, b1, (short)0, acc1, false, false);
    acc2 = __builtin_amdgcn_wmma_f32_16x16x32_bf16(false, a, false, b2, (short)0, acc2, false, false);
    acc3 = __builtin_amdgcn_wmma_f32_16x16x32_bf16(false, a, false, b3, (short)0, acc3, false, false);
  }
  v8f accs[4] = {acc0, acc1, acc2, acc3};
#pragma unroll
  for (int q = 0; q < 4; q++) {
    int col = tileN + q * 16 + laneL;
    float bias = bq[col];
#pragma unroll
    for (int j = 0; j < 8; j++) {
      int row = tileM + j + hi * 8;   // C layout: VGPR j -> M=j (lanes 0-15) / M=j+8
      qraw[(size_t)row * HQDIM + col] = accs[q][j] + bias;
    }
  }
}

// ---------------- batchnorm stats: partial sums over row chunks, then finalize ----------------
__global__ void k_bn_partial(const float* __restrict__ qraw,
                             float* __restrict__ psum, float* __restrict__ psumsq) {
  int col = blockIdx.x * blockDim.x + threadIdx.x;
  int chunk = blockIdx.y;
  float s = 0.f, ss = 0.f;
  int r0 = chunk * (TDIM / BNCHUNK), r1 = r0 + (TDIM / BNCHUNK);
  for (int r = r0; r < r1; r++) {
    float v = qraw[(size_t)r * HQDIM + col];   // coalesced across cols
    s += v; ss += v * v;
  }
  psum[(size_t)chunk * HQDIM + col] = s;
  psumsq[(size_t)chunk * HQDIM + col] = ss;
}

__global__ void k_bn_finalize(const float* __restrict__ psum,
                              const float* __restrict__ psumsq,
                              const float* __restrict__ gamma,
                              const float* __restrict__ beta,
                              float* __restrict__ scale,
                              float* __restrict__ shift) {
  int col = blockIdx.x * blockDim.x + threadIdx.x;
  if (col >= HQDIM) return;
  float s = 0.f, ss = 0.f;
#pragma unroll
  for (int c = 0; c < BNCHUNK; c++) {
    s  += psum[(size_t)c * HQDIM + col];
    ss += psumsq[(size_t)c * HQDIM + col];
  }
  float mu = s * (1.f / TDIM);
  float var = ss * (1.f / TDIM) - mu * mu;     // biased variance
  float sc = rsqrtf(var + 1e-5f) * gamma[col];
  scale[col] = sc;
  shift[col] = beta[col] - mu * sc;
}

__global__ void k_bn_apply(const float* __restrict__ qraw,
                           const float* __restrict__ scale,
                           const float* __restrict__ shift,
                           unsigned short* __restrict__ qb) {
  int i = blockIdx.x * blockDim.x + threadIdx.x;
  if (i >= TDIM * HQDIM) return;
  int col = i & (HQDIM - 1);
  qb[i] = f32_to_bf16_rne(fmaf(qraw[i], scale[col], shift[col]));
}

// ------- sub-key score GEMMs, 16x64 strip per wave, batched over (head, half) -------
// s layout: [T][H][2][NSUB]
__global__ void k_gemm_scores(const unsigned short* __restrict__ qb,
                              const unsigned short* __restrict__ c1b,
                              const unsigned short* __restrict__ c2b,
                              float* __restrict__ s) {
  int wid = threadIdx.x >> 5;
  int lane = threadIdx.x & 31;
  int laneL = lane & 15, hi = lane >> 4;
  int hp = blockIdx.z;                 // h*2 + p
  int h = hp >> 1, p = hp & 1;
  int tileN = (blockIdx.x * 4 + wid) * 64;
  int tileM = blockIdx.y * 16;
  const unsigned short* A = qb + h * QDIM + p * QHALF;                   // lda = HQDIM
  const unsigned short* Bt = (p ? c2b : c1b) + (size_t)h * NSUB * QHALF; // [NSUB][QHALF]
  v8f acc0 = {}, acc1 = {}, acc2 = {}, acc3 = {};
  for (int k0 = 0; k0 < QHALF; k0 += 32) {
    v16bf a  = load_a_frag(A, HQDIM, tileM + laneL, k0, hi);
    v16bf b0 = load_b_frag(Bt, QHALF, tileN +  0 + laneL, k0, hi);
    v16bf b1 = load_b_frag(Bt, QHALF, tileN + 16 + laneL, k0, hi);
    v16bf b2 = load_b_frag(Bt, QHALF, tileN + 32 + laneL, k0, hi);
    v16bf b3 = load_b_frag(Bt, QHALF, tileN + 48 + laneL, k0, hi);
    acc0 = __builtin_amdgcn_wmma_f32_16x16x32_bf16(false, a, false, b0, (short)0, acc0, false, false);
    acc1 = __builtin_amdgcn_wmma_f32_16x16x32_bf16(false, a, false, b1, (short)0, acc1, false, false);
    acc2 = __builtin_amdgcn_wmma_f32_16x16x32_bf16(false, a, false, b2, (short)0, acc2, false, false);
    acc3 = __builtin_amdgcn_wmma_f32_16x16x32_bf16(false, a, false, b3, (short)0, acc3, false, false);
  }
  float* C = s + (size_t)hp * NSUB;    // + row * (H*2*NSUB)
  v8f accs[4] = {acc0, acc1, acc2, acc3};
#pragma unroll
  for (int q = 0; q < 4; q++) {
    int col = tileN + q * 16 + laneL;
#pragma unroll
    for (int j = 0; j < 8; j++) {
      int row = tileM + j + hi * 8;
      C[(size_t)row * (NHEAD * 2 * NSUB) + col] = accs[q][j];
    }
  }
}

// ---------------- top-32 of 512 per (t,h,half): one wave per row ----------------
__global__ void k_topk(const float* __restrict__ s,
                       float* __restrict__ topv, int* __restrict__ topi) {
  int wid = threadIdx.x >> 5;
  int lane = threadIdx.x & 31;
  int row = blockIdx.x * 4 + wid;               // over [T][H][2]
  const float* src = s + (size_t)row * NSUB;
  float local[16];
#pragma unroll
  for (int i = 0; i < 16; i++) local[i] = src[i * 32 + lane];
  float rv = -INFINITY; int ri = 0;
  for (int kk = 0; kk < KSEL; kk++) {
    float bestv = -INFINITY; int bestslot = 0;
#pragma unroll
    for (int i = 0; i < 16; i++)
      if (local[i] > bestv) { bestv = local[i]; bestslot = i; }
    float v = bestv;
    int pos = bestslot * 32 + lane;             // global index n in [0,512)
    for (int off = 16; off; off >>= 1) {        // wave32 argmax
      float ov = __shfl_xor(v, off, 32);
      int   op = __shfl_xor(pos, off, 32);
      if (ov > v || (ov == v && op < pos)) { v = ov; pos = op; }
    }
    if ((pos & 31) == lane) local[pos >> 5] = -INFINITY;  // mask winner
    if (kk == lane) { rv = v; ri = pos; }
  }
  topv[(size_t)row * KSEL + lane] = rv;
  topi[(size_t)row * KSEL + lane] = ri;
}

// ---------- cartesian merge (32x32 -> top-32) + softmax + flat index ----------
__global__ void k_combine(const float* __restrict__ topv,
                          const int* __restrict__ topi,
                          float* __restrict__ attn, int* __restrict__ memidx) {
  int wid = threadIdx.x >> 5;
  int lane = threadIdx.x & 31;
  int row = blockIdx.x * 4 + wid;               // t*NHEAD + h
  const float* tv1 = topv + (size_t)(row * 2 + 0) * KSEL;
  const float* tv2 = topv + (size_t)(row * 2 + 1) * KSEL;
  const int*   ti1 = topi + (size_t)(row * 2 + 0) * KSEL;
  const int*   ti2 = topi + (size_t)(row * 2 + 1) * KSEL;
  float t1 = tv1[lane];
  float t2 = tv2[lane];
  unsigned used = 0u;
  float rv = -INFINITY; int rpos = 0;
  for (int kk = 0; kk < KSEL; kk++) {
    float bestv = -INFINITY; int bestj = 0;
#pragma unroll
    for (int j = 0; j < 32; j++) {
      float c = t1 + __shfl(t2, j, 32);         // combo (lane, j)
      bool avail = ((used >> j) & 1u) == 0u;
      if (avail && c > bestv) { bestv = c; bestj = j; }
    }
    float v = bestv;
    int pos = (lane << 5) | bestj;
    for (int off = 16; off; off >>= 1) {
      float ov = __shfl_xor(v, off, 32);
      int   op = __shfl_xor(pos, off, 32);
      if (ov > v || (ov == v && op < pos)) { v = ov; pos = op; }
    }
    if ((pos >> 5) == lane) used |= (1u << (pos & 31));
    if (kk == lane) { rv = v; rpos = pos; }
  }
  // softmax over 32 selected; selection is descending so lane 0 holds the max
  float m = __shfl(rv, 0, 32);
  float e = __expf(rv - m);
  float ssum = e;
  for (int off = 16; off; off >>= 1) ssum += __shfl_xor(ssum, off, 32);
  float a = e / ssum;
  int i1 = ti1[rpos >> 5];
  int i2 = ti2[rpos & 31];
  attn[(size_t)row * KSEL + lane] = a;
  memidx[(size_t)row * KSEL + lane] = i1 * NSUB + i2;
}

// ------- bandwidth-critical stage: gather 128 value rows per token, reduce -------
// 128 threads/token, float4 per lane -> one global_load_b128 covers the 2KB row.
__global__ void k_gather(const float* __restrict__ attn,
                         const int* __restrict__ memidx,
                         const float* __restrict__ values,
                         float* __restrict__ out) {
  __shared__ float w[NHEAD * KSEL];
  __shared__ int   mi[NHEAD * KSEL];
  int t = blockIdx.x;
  int tid = threadIdx.x;                        // 0..127, 4 d-columns per lane
  w[tid]  = attn[(size_t)t * NHEAD * KSEL + tid];
  mi[tid] = memidx[(size_t)t * NHEAD * KSEL + tid];
  __syncthreads();
  float4 acc = make_float4(0.f, 0.f, 0.f, 0.f);
#pragma unroll 4
  for (int e = 0; e < NHEAD * KSEL; e++) {
    if (e + 1 < NHEAD * KSEL)                   // keep HBM pipe fed
      __builtin_prefetch(values + (size_t)mi[e + 1] * DMODEL + tid * 4, 0, 0);
    const float4* row = (const float4*)(values + (size_t)mi[e] * DMODEL);
    float4 v = row[tid];
    float we = w[e];
    acc.x = fmaf(we, v.x, acc.x);
    acc.y = fmaf(we, v.y, acc.y);
    acc.z = fmaf(we, v.z, acc.z);
    acc.w = fmaf(we, v.w, acc.w);
  }
  ((float4*)(out + (size_t)t * DMODEL))[tid] = acc;
}

extern "C" void kernel_launch(void* const* d_in, const int* in_sizes, int n_in,
                              void* d_out, int out_size, void* d_ws, size_t ws_size,
                              hipStream_t stream) {
  const float* x      = (const float*)d_in[0];
  const float* Wq     = (const float*)d_in[1];
  const float* bq     = (const float*)d_in[2];
  const float* gamma  = (const float*)d_in[3];
  const float* beta   = (const float*)d_in[4];
  const float* c1     = (const float*)d_in[5];
  const float* c2     = (const float*)d_in[6];
  const float* values = (const float*)d_in[7];
  float* out = (float*)d_out;

  // workspace carve-out (~68 MB total), 256B aligned slices
  char* ws = (char*)d_ws;
  size_t off = 0;
  auto take = [&](size_t bytes) -> char* {
    char* p = ws + off;
    off = (off + bytes + 255) & ~(size_t)255;
    return p;
  };
  unsigned short* xb    = (unsigned short*)take((size_t)TDIM * DMODEL * 2);
  unsigned short* wbT   = (unsigned short*)take((size_t)HQDIM * DMODEL * 2);
  unsigned short* c1b   = (unsigned short*)take((size_t)NHEAD * NSUB * QHALF * 2);
  unsigned short* c2b   = (unsigned short*)take((size_t)NHEAD * NSUB * QHALF * 2);
  float*          qraw  = (float*)take((size_t)TDIM * HQDIM * 4);
  float*          psum  = (float*)take((size_t)BNCHUNK * HQDIM * 4);
  float*          psumsq= (float*)take((size_t)BNCHUNK * HQDIM * 4);
  float*          scale = (float*)take((size_t)HQDIM * 4);
  float*          shift = (float*)take((size_t)HQDIM * 4);
  unsigned short* qb    = (unsigned short*)take((size_t)TDIM * HQDIM * 2);
  float*          s     = (float*)take((size_t)TDIM * NHEAD * 2 * NSUB * 4);
  float*          topv  = (float*)take((size_t)TDIM * NHEAD * 2 * KSEL * 4);
  int*            topi  = (int*)take((size_t)TDIM * NHEAD * 2 * KSEL * 4);
  float*          attn  = (float*)take((size_t)TDIM * NHEAD * KSEL * 4);
  int*            memidx= (int*)take((size_t)TDIM * NHEAD * KSEL * 4);

  int n;
  n = TDIM * DMODEL;
  k_convert_bf16<<<(n + 255) / 256, 256, 0, stream>>>(x, xb, n);
  n = DMODEL * HQDIM;
  k_transpose_bf16<<<(n + 255) / 256, 256, 0, stream>>>(Wq, wbT);
  n = NHEAD * NSUB * QHALF;
  k_convert_bf16<<<(n + 255) / 256, 256, 0, stream>>>(c1, c1b, n);
  k_convert_bf16<<<(n + 255) / 256, 256, 0, stream>>>(c2, c2b, n);

  k_gemm_q<<<dim3(HQDIM / 256, TDIM / 16), 128, 0, stream>>>(xb, wbT, bq, qraw);
  k_bn_partial<<<dim3(HQDIM / 256, BNCHUNK), 256, 0, stream>>>(qraw, psum, psumsq);
  k_bn_finalize<<<HQDIM / 256, 256, 0, stream>>>(psum, psumsq, gamma, beta, scale, shift);
  n = TDIM * HQDIM;
  k_bn_apply<<<(n + 255) / 256, 256, 0, stream>>>(qraw, scale, shift, qb);
  k_gemm_scores<<<dim3(NSUB / 256, TDIM / 16, NHEAD * 2), 128, 0, stream>>>(qb, c1b, c2b, s);
  k_topk<<<(TDIM * NHEAD * 2) / 4, 128, 0, stream>>>(s, topv, topi);
  k_combine<<<(TDIM * NHEAD) / 4, 128, 0, stream>>>(topv, topi, attn, memidx);
  k_gather<<<TDIM, NHEAD * KSEL, 0, stream>>>(attn, memidx, values, out);

  (void)in_sizes; (void)n_in; (void)out_size; (void)ws_size;
}
